// AtomTransformer_85478439125475
// MI455X (gfx1250) — matrix operand order repaired and checked by reading, not compiled, not run on hardware
//
#include <hip/hip_runtime.h>
#include <hip/hip_bf16.h>

typedef __attribute__((ext_vector_type(16))) _Float16 v16h;
typedef __attribute__((ext_vector_type(8)))  float    v8f;
typedef __attribute__((ext_vector_type(4)))  unsigned int v4u;

union H16 { v16h v; v4u u[2]; _Float16 e[16]; };
union H8  { v4u u; _Float16 e[8]; };

static __device__ __forceinline__ v8f vzero8() {
    v8f z = {0.f,0.f,0.f,0.f,0.f,0.f,0.f,0.f};
    return z;
}

static __device__ __forceinline__ v8f wmma16(v16h a, v16h b, v8f c) {
    // D = A(16x32 f16) * B(32x16 f16) + C(16x16 f32)
    return __builtin_amdgcn_wmma_f32_16x16x32_f16(false, a, false, b, (short)0, c, false, false);
}

// Contiguous 16-half load (B-matrix fragment: b[e] = row[base+e], lane holds
// col = L&15, K-subrange selected by caller via base offset with g = L>>4).
static __device__ __forceinline__ v16h load16h(const _Float16* p) {
    H16 h;
    const v4u* q = (const v4u*)p;
    h.u[0] = q[0];
    h.u[1] = q[1];
    return h.v;
}

// A-matrix fragment for 16x32 f16: lane L (m=L&15, g=L>>4):
//   K = kb + g*16 + e (contiguous, matches B fragment)
static __device__ __forceinline__ v16h loadA(const _Float16* rowp, int kb, int g) {
    H16 h;
    h.u[0] = *(const v4u*)(rowp + kb + g*16);
    h.u[1] = *(const v4u*)(rowp + kb + g*16 + 8);
    return h.v;
}

// ---------------------------------------------------------------- transpose+cvt
__global__ __launch_bounds__(256) void tcvt_kernel(const float* __restrict__ src,
                                                   _Float16* __restrict__ dst,
                                                   int K, int N) {
    int idx = blockIdx.x * 256 + threadIdx.x;
    if (idx >= K * N) return;
    int n = idx / K;
    int k = idx - n * K;
    dst[idx] = (_Float16)src[(size_t)k * N + n];   // dst[n][k] = src[k][n]
}

// ---------------------------------------------------------------- LN + Q/K/V/gate
__global__ __launch_bounds__(128) void ln_proj_kernel(
    const float* __restrict__ qsrc, const float* __restrict__ g, const float* __restrict__ b,
    const _Float16* __restrict__ WqT, const _Float16* __restrict__ WkT,
    const _Float16* __restrict__ WvT, const _Float16* __restrict__ WgT,
    const float* __restrict__ bq,
    _Float16* __restrict__ qf, _Float16* __restrict__ kf,
    _Float16* __restrict__ vT, float* __restrict__ gate, int Nrow) {
    __shared__ __align__(16) _Float16 xs[16][128];
    const int t = threadIdx.x;
    const int strip = blockIdx.x;
    {   // layernorm of 16 rows, 8 lanes/row
        int r16 = t >> 3, c0 = (t & 7) * 16;
        const float* rp = qsrc + ((size_t)(strip*16 + r16)) * 128 + c0;
        float v[16]; float s = 0.f, s2 = 0.f;
        for (int j = 0; j < 16; ++j) { v[j] = rp[j]; s += v[j]; s2 += v[j]*v[j]; }
        for (int m = 1; m < 8; m <<= 1) { s += __shfl_xor(s, m, 32); s2 += __shfl_xor(s2, m, 32); }
        float mu = s * (1.f/128.f);
        float var = s2 * (1.f/128.f) - mu*mu;
        float rstd = rsqrtf(var + 1e-5f);
        for (int j = 0; j < 16; ++j) {
            int c = c0 + j;
            xs[r16][c] = (_Float16)((v[j]-mu)*rstd*g[c] + b[c]);
        }
    }
    __syncthreads();
    const int w = t >> 5, L = t & 31, nn = L & 15, gg = L >> 4;
    const _Float16* WT = (w==0) ? WqT : (w==1) ? WkT : (w==2) ? WvT : WgT;
    v8f acc[8];
    for (int j = 0; j < 8; ++j) acc[j] = vzero8();
    for (int kc = 0; kc < 4; ++kc) {
        v16h a = loadA(&xs[nn][0], kc*32, gg);
        for (int ct = 0; ct < 8; ++ct) {
            v16h bm = load16h(WT + (size_t)(ct*16 + nn)*128 + kc*32 + gg*16);
            acc[ct] = wmma16(a, bm, acc[ct]);
        }
    }
    for (int ct = 0; ct < 8; ++ct) {
        int col = ct*16 + nn;
        if (w == 2) {  // V transposed: contiguous 8-half pack per lane
            H8 pk;
            for (int r = 0; r < 8; ++r) pk.e[r] = (_Float16)acc[ct][r];
            *(v4u*)(vT + (size_t)col*Nrow + strip*16 + gg*8) = pk.u;
        } else {
            for (int r = 0; r < 8; ++r) {
                int row = strip*16 + r + 8*gg;
                float val = acc[ct][r];
                if (w == 0)      qf[(size_t)row*128 + col] = (_Float16)(val + bq[col]);
                else if (w == 1) kf[(size_t)row*128 + col] = (_Float16)val;
                else             gate[(size_t)row*128 + col] = 1.f/(1.f + expf(-val));
            }
        }
    }
}

// ---------------------------------------------------------------- windowed attention + gate + Wo + residual
__global__ __launch_bounds__(128) void attn_kernel(
    const _Float16* __restrict__ qf, const _Float16* __restrict__ kf,
    const _Float16* __restrict__ vT, const float* __restrict__ gate,
    const float* __restrict__ plm,
    const float* __restrict__ png, const float* __restrict__ pnb,
    const float* __restrict__ wpb, const _Float16* __restrict__ WoT,
    const float* __restrict__ qsrc, float* __restrict__ qdst, int Nrow) {
    __shared__ __align__(16) float lg[4][32][128];   // 64 KB logits / later f16 overlays
    const int t = threadIdx.x, ci = blockIdx.x;
    const int qb0 = ci * 32;
    int ks = ci*32 - 48; if (ks < 0) ks = 0;
    int ke = ci*32 + 80; if (ke < 128) ke = 128; if (ke > Nrow) ke = Nrow;
    const int width = ke - ks;
    // ---- phase 1: pair bias (LN over 16 chans, x Wpb) into LDS; mask tail
    {
        const int kk = t;
        const bool valid = kk < width;
        for (int qr = 0; qr < 32; ++qr) {
            if (valid) {
                const float* pr = plm + ((size_t)(qb0 + qr) * Nrow + (ks + kk)) * 16;
                float x[16]; float s = 0.f;
                for (int c = 0; c < 16; ++c) { x[c] = pr[c]; s += x[c]; }
                float mu = s * (1.f/16.f);
                float s2 = 0.f;
                for (int c = 0; c < 16; ++c) { float d = x[c]-mu; s2 += d*d; }
                float rstd = rsqrtf(s2 * (1.f/16.f) + 1e-5f);
                float b0=0.f,b1=0.f,b2=0.f,b3=0.f;
                for (int c = 0; c < 16; ++c) {
                    float y = (x[c]-mu)*rstd*png[c] + pnb[c];
                    b0 += y * wpb[c*4+0];
                    b1 += y * wpb[c*4+1];
                    b2 += y * wpb[c*4+2];
                    b3 += y * wpb[c*4+3];
                }
                lg[0][qr][kk]=b0; lg[1][qr][kk]=b1; lg[2][qr][kk]=b2; lg[3][qr][kk]=b3;
            } else {
                lg[0][qr][kk]=-1e10f; lg[1][qr][kk]=-1e10f; lg[2][qr][kk]=-1e10f; lg[3][qr][kk]=-1e10f;
            }
        }
    }
    __syncthreads();
    const int h = t >> 5, L = t & 31, nn = L & 15, gg = L >> 4;
    const float scale = 0.17677669529663687f;  // 1/sqrt(32)
    // ---- phase 2: logits += scale * q.kT  (one WMMA per 16x16 tile, K=C_HEAD=32)
    for (int mt = 0; mt < 2; ++mt) {
        v16h a = loadA(qf + (size_t)(qb0 + mt*16 + nn) * 128, h*32, gg);
        for (int nt = 0; nt < 8; ++nt) {
            int key = ks + nt*16 + nn; if (key > Nrow-1) key = Nrow-1;
            v16h bm = load16h(kf + (size_t)key * 128 + h*32 + gg*16);
            v8f acc = wmma16(a, bm, vzero8());
            for (int r = 0; r < 8; ++r)
                lg[h][mt*16 + r + 8*gg][nt*16 + nn] += acc[r] * scale;
        }
    }
    __syncthreads();
    // ---- phase 3: softmax per row, write f16 probabilities overlaid in place
    {
        const int hh = t >> 5, qr = t & 31;
        float* row = &lg[hh][qr][0];
        float mx = -1e30f;
        for (int j = 0; j < 128; ++j) mx = fmaxf(mx, row[j]);
        float s = 0.f;
        for (int j = 0; j < 128; ++j) s += expf(row[j] - mx);
        float inv = 1.f / s;
        _Float16* hr = (_Float16*)row;
        for (int j = 0; j < 128; ++j) hr[j] = (_Float16)(expf(row[j] - mx) * inv);
    }
    __syncthreads();
    // ---- phase 4: attn = w @ v
    const bool tail = (ks + 128 > Nrow);
    v8f at[2][2];
    for (int a0 = 0; a0 < 2; ++a0) for (int a1 = 0; a1 < 2; ++a1) at[a0][a1] = vzero8();
    for (int mt = 0; mt < 2; ++mt) {
        for (int kc = 0; kc < 4; ++kc) {
            v16h a = loadA((const _Float16*)&lg[h][mt*16 + nn][0], kc*32, gg);
            for (int nt = 0; nt < 2; ++nt) {
                int col = h*32 + nt*16 + nn;
                v16h bm;
                if (!tail) {
                    bm = load16h(vT + (size_t)col * Nrow + ks + kc*32 + gg*16);
                } else {
                    H16 hb;
                    for (int e = 0; e < 16; ++e) {
                        int key = ks + kc*32 + gg*16 + e; if (key > Nrow-1) key = Nrow-1;
                        hb.e[e] = vT[(size_t)col * Nrow + key];
                    }
                    bm = hb.v;
                }
                at[mt][nt] = wmma16(a, bm, at[mt][nt]);
            }
        }
    }
    __syncthreads();
    // ---- gate * attn -> f16 overlay (reuse lg)
    _Float16* a2 = (_Float16*)&lg[0][0][0];
    for (int mt = 0; mt < 2; ++mt)
        for (int nt = 0; nt < 2; ++nt) {
            int col = h*32 + nt*16 + nn;
            for (int r = 0; r < 8; ++r) {
                int m = mt*16 + r + 8*gg;
                float gv = gate[(size_t)(qb0 + m) * 128 + col];
                a2[m*128 + col] = (_Float16)(gv * at[mt][nt][r]);
            }
        }
    __syncthreads();
    // ---- phase 5: (gate*attn) @ Wo + residual
    v8f oc[2][2];
    for (int a0 = 0; a0 < 2; ++a0) for (int a1 = 0; a1 < 2; ++a1) oc[a0][a1] = vzero8();
    for (int kc = 0; kc < 4; ++kc) {
        v16h aA = loadA(a2 + (size_t)(0*16 + nn) * 128, kc*32, gg);
        v16h aB = loadA(a2 + (size_t)(1*16 + nn) * 128, kc*32, gg);
        for (int cti = 0; cti < 2; ++cti) {
            int ct = h*2 + cti;
            v16h bm = load16h(WoT + (size_t)(ct*16 + nn)*128 + kc*32 + gg*16);
            oc[0][cti] = wmma16(aA, bm, oc[0][cti]);
            oc[1][cti] = wmma16(aB, bm, oc[1][cti]);
        }
    }
    for (int mt = 0; mt < 2; ++mt)
        for (int cti = 0; cti < 2; ++cti) {
            int col = (h*2 + cti)*16 + nn;
            for (int r = 0; r < 8; ++r) {
                int row = qb0 + mt*16 + r + 8*gg;
                qdst[(size_t)row*128 + col] = qsrc[(size_t)row*128 + col] + oc[mt][cti][r];
            }
        }
}

// ---------------------------------------------------------------- transition MLP
__global__ __launch_bounds__(128) void transition_kernel(
    const float* __restrict__ qin, const float* __restrict__ g, const float* __restrict__ b,
    const _Float16* __restrict__ W1T, const float* __restrict__ bt1,
    const _Float16* __restrict__ W2T, const float* __restrict__ bt2,
    float* __restrict__ qout) {
    __shared__ __align__(16) _Float16 ts[16][128];
    __shared__ __align__(16) _Float16 h1s[16][512];
    const int t = threadIdx.x;
    const int strip = blockIdx.x;
    {
        int r16 = t >> 3, c0 = (t & 7) * 16;
        const float* rp = qin + ((size_t)(strip*16 + r16)) * 128 + c0;
        float v[16]; float s = 0.f, s2 = 0.f;
        for (int j = 0; j < 16; ++j) { v[j] = rp[j]; s += v[j]; s2 += v[j]*v[j]; }
        for (int m = 1; m < 8; m <<= 1) { s += __shfl_xor(s, m, 32); s2 += __shfl_xor(s2, m, 32); }
        float mu = s * (1.f/128.f);
        float var = s2 * (1.f/128.f) - mu*mu;
        float rstd = rsqrtf(var + 1e-5f);
        for (int j = 0; j < 16; ++j) {
            int c = c0 + j;
            ts[r16][c] = (_Float16)((v[j]-mu)*rstd*g[c] + b[c]);
        }
    }
    __syncthreads();
    const int w = t >> 5, L = t & 31, nn = L & 15, gg = L >> 4;
    v8f acc[8];
    for (int j = 0; j < 8; ++j) acc[j] = vzero8();
    for (int kc = 0; kc < 4; ++kc) {
        v16h a = loadA(&ts[nn][0], kc*32, gg);
        for (int j = 0; j < 8; ++j) {
            int ct = w*8 + j;
            v16h bm = load16h(W1T + (size_t)(ct*16 + nn)*128 + kc*32 + gg*16);
            acc[j] = wmma16(a, bm, acc[j]);
        }
    }
    for (int j = 0; j < 8; ++j) {
        int col = (w*8 + j)*16 + nn;
        float bb = bt1[col];
        for (int r = 0; r < 8; ++r) {
            float v = acc[j][r] + bb;
            h1s[r + 8*gg][col] = (_Float16)(v > 0.f ? v : 0.f);
        }
    }
    __syncthreads();
    v8f o[2];
    o[0] = vzero8(); o[1] = vzero8();
    for (int kc = 0; kc < 16; ++kc) {
        v16h a = loadA(&h1s[nn][0], kc*32, gg);
        for (int j = 0; j < 2; ++j) {
            int ct = w*2 + j;
            v16h bm = load16h(W2T + (size_t)(ct*16 + nn)*512 + kc*32 + gg*16);
            o[j] = wmma16(a, bm, o[j]);
        }
    }
    for (int j = 0; j < 2; ++j) {
        int col = (w*2 + j)*16 + nn;
        for (int r = 0; r < 8; ++r) {
            int row = strip*16 + r + 8*gg;
            qout[(size_t)row*128 + col] = qin[(size_t)row*128 + col] + o[j][r] + bt2[col];
        }
    }
}

// ---------------------------------------------------------------- host launcher
extern "C" void kernel_launch(void* const* d_in, const int* in_sizes, int n_in,
                              void* d_out, int out_size, void* d_ws, size_t ws_size,
                              hipStream_t stream) {
    (void)n_in; (void)out_size; (void)ws_size;
    const float* ql    = (const float*)d_in[0];
    // d_in[1] = cl, unused by the reference
    const float* plm   = (const float*)d_in[2];
    const float* lnq_g = (const float*)d_in[3];
    const float* lnq_b = (const float*)d_in[4];
    const float* lnp_g = (const float*)d_in[5];
    const float* lnp_b = (const float*)d_in[6];
    const float* Wq    = (const float*)d_in[7];
    const float* bq    = (const float*)d_in[8];
    const float* Wk    = (const float*)d_in[9];
    const float* Wv    = (const float*)d_in[10];
    const float* Wpb   = (const float*)d_in[11];
    const float* Wg    = (const float*)d_in[12];
    const float* Wo    = (const float*)d_in[13];
    const float* lnt_g = (const float*)d_in[14];
    const float* lnt_b = (const float*)d_in[15];
    const float* Wt1   = (const float*)d_in[16];
    const float* bt1   = (const float*)d_in[17];
    const float* Wt2   = (const float*)d_in[18];
    const float* bt2   = (const float*)d_in[19];

    const int Nrow = in_sizes[0] / 128;   // 2048

    char* p = (char*)d_ws;
    float*    qlA  = (float*)p;            p += (size_t)Nrow * 128 * 4;
    _Float16* qf   = (_Float16*)p;         p += (size_t)Nrow * 128 * 2;
    _Float16* kf   = (_Float16*)p;         p += (size_t)Nrow * 128 * 2;
    _Float16* vT   = (_Float16*)p;         p += (size_t)Nrow * 128 * 2;
    float*    gate = (float*)p;            p += (size_t)Nrow * 128 * 4;
    _Float16* WqT  = (_Float16*)p;         p += 128*128*2;
    _Float16* WkT  = (_Float16*)p;         p += 128*128*2;
    _Float16* WvT  = (_Float16*)p;         p += 128*128*2;
    _Float16* WgT  = (_Float16*)p;         p += 128*128*2;
    _Float16* WoT  = (_Float16*)p;         p += 128*128*2;
    _Float16* W1T  = (_Float16*)p;         p += 128*512*2;
    _Float16* W2T  = (_Float16*)p;         p += 512*128*2;

    const int strips = Nrow / 16;
    const int qblocks = Nrow / 32;

    for (int i = 0; i < 3; ++i) {
        const float* src = (i == 0) ? ql : qlA;
        float* fin = (i == 2) ? (float*)d_out : qlA;
        // f16 transposed weights for WMMA B-fragment loads
        tcvt_kernel<<<64,  256, 0, stream>>>(Wq  + (size_t)i*16384, WqT, 128, 128);
        tcvt_kernel<<<64,  256, 0, stream>>>(Wk  + (size_t)i*16384, WkT, 128, 128);
        tcvt_kernel<<<64,  256, 0, stream>>>(Wv  + (size_t)i*16384, WvT, 128, 128);
        tcvt_kernel<<<64,  256, 0, stream>>>(Wg  + (size_t)i*16384, WgT, 128, 128);
        tcvt_kernel<<<64,  256, 0, stream>>>(Wo  + (size_t)i*16384, WoT, 128, 128);
        tcvt_kernel<<<256, 256, 0, stream>>>(Wt1 + (size_t)i*65536, W1T, 128, 512);
        tcvt_kernel<<<256, 256, 0, stream>>>(Wt2 + (size_t)i*65536, W2T, 512, 128);

        ln_proj_kernel<<<strips, 128, 0, stream>>>(
            src, lnq_g + i*128, lnq_b + i*128,
            WqT, WkT, WvT, WgT, bq + i*128,
            qf, kf, vT, gate, Nrow);

        attn_kernel<<<qblocks, 128, 0, stream>>>(
            qf, kf, vT, gate, plm,
            lnp_g + i*16, lnp_b + i*16, Wpb + i*64,
            WoT, src, qlA, Nrow);

        transition_kernel<<<strips, 128, 0, stream>>>(
            qlA, lnt_g + i*128, lnt_b + i*128,
            W1T, bt1 + i*512, W2T, bt2 + i*128, fin);
    }
}
